// EdgeModel_39702677684918
// MI455X (gfx1250) — matrix operand order repaired and checked by reading, not compiled, not run on hardware
//
#include <hip/hip_runtime.h>

#define DH   128
#define NRBF 16
#define NL   3

typedef __attribute__((ext_vector_type(16))) __bf16        v16bf;
typedef __attribute__((ext_vector_type(8)))  float         v8f;
typedef __attribute__((ext_vector_type(4)))  unsigned int  u32x4;

__device__ __forceinline__ unsigned short f2bf(float f) {
  unsigned u = __builtin_bit_cast(unsigned, f);
  unsigned r = u + 0x7FFFu + ((u >> 16) & 1u);
  return (unsigned short)(r >> 16);
}

// Build a 16x32 (A) or 32x16 (B) bf16 fragment from two 16-byte runs.
__device__ __forceinline__ v16bf frag2(const unsigned short* p0, const unsigned short* p1) {
  union { u32x4 u[2]; v16bf v; } f;
  f.u[0] = *(const u32x4*)p0;
  f.u[1] = *(const u32x4*)p1;
  return f.v;
}

// ---------------------------------------------------------------------------
// Pack fp32 weights [K][128] -> bf16 fragment-native layout [kk][n][32]
// (32 consecutive K values per (kstep, col), zero padded past Ksrc).
// ---------------------------------------------------------------------------
__global__ __launch_bounds__(256) void pack_kernel(const float* __restrict__ W,
                                                   unsigned short* __restrict__ dst,
                                                   int Ksrc, int total) {
  int idx = blockIdx.x * 256 + threadIdx.x;
  if (idx >= total) return;
  int kk  = idx >> 12;          // / (128*32)
  int rem = idx & 4095;
  int n   = rem >> 5;
  int kv  = rem & 31;
  int k   = kk * 32 + kv;
  dst[idx] = (k < Ksrc) ? f2bf(W[(size_t)k * DH + n]) : (unsigned short)0;
}

__global__ __launch_bounds__(256) void zero_kernel(float* __restrict__ p, int n) {
  int i = blockIdx.x * 256 + threadIdx.x;
  if (i < n) p[i] = 0.f;
}

// ---------------------------------------------------------------------------
// Node embedding MLP: coords(2) -> 128 -> 128. Tiny K, done on VALU.
// ---------------------------------------------------------------------------
__global__ __launch_bounds__(128) void node_kernel(const float* __restrict__ coords,
                                                   const float* __restrict__ W1,
                                                   const float* __restrict__ b1,
                                                   const float* __restrict__ W2,
                                                   const float* __restrict__ b2,
                                                   float* __restrict__ h_f32,
                                                   unsigned short* __restrict__ h_bf,
                                                   int N) {
  int n = blockIdx.x;
  if (n >= N) return;
  int j = threadIdx.x;
  __shared__ float hid[DH];
  float c0 = coords[2 * n], c1 = coords[2 * n + 1];
  float v = fmaf(c0, W1[j], fmaf(c1, W1[DH + j], b1[j]));
  hid[j] = fmaxf(v, 0.f);
  __syncthreads();
  float o = b2[j];
#pragma unroll 8
  for (int k = 0; k < DH; ++k) o = fmaf(hid[k], W2[k * DH + j], o);
  h_f32[(size_t)n * DH + j] = o;
  h_bf [(size_t)n * DH + j] = f2bf(o);
}

// ---------------------------------------------------------------------------
// Message MLP (per layer): concat[h[src],h[dst],rbf] (272->pad288) -> relu 128
// -> 128, then atomic scatter-add into agg[dst]. 64 edges x 128 cols / block.
// ---------------------------------------------------------------------------
__global__ __launch_bounds__(256) void msg_kernel(const unsigned short* __restrict__ h_bf,
                                                  const float* __restrict__ coords,
                                                  const int* __restrict__ srcI,
                                                  const int* __restrict__ dstI,
                                                  const unsigned short* __restrict__ W1p,
                                                  const float* __restrict__ b1,
                                                  const unsigned short* __restrict__ W2p,
                                                  const float* __restrict__ b2,
                                                  float* __restrict__ agg,
                                                  int E) {
  __shared__ alignas(16) unsigned short A1[64 * 296];  // stride 296 (16B aligned rows)
  __shared__ alignas(16) unsigned short A2[64 * 136];
  __shared__ int   s_src[64];
  __shared__ int   s_dst[64];
  __shared__ float s_r[64];

  const int tid = threadIdx.x;
  const int e0  = blockIdx.x * 64;

  if (tid < 64) {
    int e = e0 + tid;
    int s = 0, d = 0; float r = 0.f;
    if (e < E) {
      s = srcI[e]; d = dstI[e];
      float dx = coords[2 * s]     - coords[2 * d];
      float dy = coords[2 * s + 1] - coords[2 * d + 1];
      r = sqrtf(dx * dx + dy * dy + 1e-8f);
    }
    s_src[tid] = s; s_dst[tid] = d; s_r[tid] = r;
  }
  __syncthreads();

  // Stage A1 = [h[src] | h[dst] | rbf | 0pad] in 16B chunks (36 per row).
  for (int idx = tid; idx < 64 * 36; idx += 256) {
    int e = idx / 36;
    int c = idx - e * 36;
    u32x4 val = {0u, 0u, 0u, 0u};
    if (e0 + e < E) {
      if (c < 16) {
        val = *(const u32x4*)(h_bf + (size_t)s_src[e] * DH + c * 8);
      } else if (c < 32) {
        val = *(const u32x4*)(h_bf + (size_t)s_dst[e] * DH + (c - 16) * 8);
      } else if (c < 34) {
        float r = s_r[e];
        int j0 = (c - 32) * 8;
        unsigned w[4];
#pragma unroll
        for (int p = 0; p < 4; ++p) {
          float d0 = (r - (float)(j0 + 2 * p)     * 0.1f) * 10.f;
          float d1 = (r - (float)(j0 + 2 * p + 1) * 0.1f) * 10.f;
          unsigned lo = f2bf(__expf(-d0 * d0));
          unsigned hi = f2bf(__expf(-d1 * d1));
          w[p] = lo | (hi << 16);
        }
        val.x = w[0]; val.y = w[1]; val.z = w[2]; val.w = w[3];
      }
    }
    *(u32x4*)&A1[e * 296 + c * 8] = val;
  }
  __syncthreads();

  const int lane = tid & 31;
  const int wv   = tid >> 5;
  const int nl   = lane & 15;
  const int half = lane >> 4;
  const int koff = half * 8;
  const int col  = wv * 16 + nl;

  v8f acc[4];
  {
    float bias = b1[col];
#pragma unroll
    for (int r = 0; r < 4; ++r)
#pragma unroll
      for (int i = 0; i < 8; ++i) acc[r][i] = bias;
  }

#pragma unroll
  for (int kk = 0; kk < 9; ++kk) {
    const unsigned short* bp = W1p + ((kk * DH + col) * 32 + half * 8);
    v16bf bf = frag2(bp, bp + 16);
#pragma unroll
    for (int r = 0; r < 4; ++r) {
      const unsigned short* ap = &A1[(r * 16 + nl) * 296 + kk * 32 + koff];
      v16bf af = frag2(ap, ap + 16);
      acc[r] = __builtin_amdgcn_wmma_f32_16x16x32_bf16(false, af, false, bf,
                                                       (short)0, acc[r], false, false);
    }
  }

#pragma unroll
  for (int r = 0; r < 4; ++r)
#pragma unroll
    for (int i = 0; i < 8; ++i) {
      int m = r * 16 + half * 8 + i;
      A2[m * 136 + col] = f2bf(fmaxf(acc[r][i], 0.f));
    }
  __syncthreads();

  v8f acc2[4];
  {
    float bias = b2[col];
#pragma unroll
    for (int r = 0; r < 4; ++r)
#pragma unroll
      for (int i = 0; i < 8; ++i) acc2[r][i] = bias;
  }

#pragma unroll
  for (int kk = 0; kk < 4; ++kk) {
    const unsigned short* bp = W2p + ((kk * DH + col) * 32 + half * 8);
    v16bf bf = frag2(bp, bp + 16);
#pragma unroll
    for (int r = 0; r < 4; ++r) {
      const unsigned short* ap = &A2[(r * 16 + nl) * 136 + kk * 32 + koff];
      v16bf af = frag2(ap, ap + 16);
      acc2[r] = __builtin_amdgcn_wmma_f32_16x16x32_bf16(false, af, false, bf,
                                                        (short)0, acc2[r], false, false);
    }
  }

#pragma unroll
  for (int r = 0; r < 4; ++r)
#pragma unroll
    for (int i = 0; i < 8; ++i) {
      int m = r * 16 + half * 8 + i;
      if (e0 + m < E) {
        __hip_atomic_fetch_add(&agg[(size_t)s_dst[m] * DH + col], acc2[r][i],
                               __ATOMIC_RELAXED, __HIP_MEMORY_SCOPE_AGENT);
      }
    }
}

// ---------------------------------------------------------------------------
// Update MLP (per layer): concat[h, agg] (256) -> relu 128 -> 128, residual.
// ---------------------------------------------------------------------------
__global__ __launch_bounds__(256) void upd_kernel(float* __restrict__ h_f32,
                                                  unsigned short* __restrict__ h_bf,
                                                  const float* __restrict__ agg,
                                                  const unsigned short* __restrict__ W1p,
                                                  const float* __restrict__ b1,
                                                  const unsigned short* __restrict__ W2p,
                                                  const float* __restrict__ b2,
                                                  int N) {
  __shared__ alignas(16) unsigned short A1[64 * 264];
  __shared__ alignas(16) unsigned short A2[64 * 136];
  const int tid = threadIdx.x;
  const int n0  = blockIdx.x * 64;

  for (int idx = tid; idx < 64 * 32; idx += 256) {
    int e = idx >> 5;
    int c = idx & 31;
    int node = n0 + e;
    u32x4 val = {0u, 0u, 0u, 0u};
    if (node < N) {
      if (c < 16) {
        val = *(const u32x4*)(h_bf + (size_t)node * DH + c * 8);
      } else {
        const float* ap = agg + (size_t)node * DH + (c - 16) * 8;
        unsigned w[4];
#pragma unroll
        for (int p = 0; p < 4; ++p) {
          unsigned lo = f2bf(ap[2 * p]);
          unsigned hi = f2bf(ap[2 * p + 1]);
          w[p] = lo | (hi << 16);
        }
        val.x = w[0]; val.y = w[1]; val.z = w[2]; val.w = w[3];
      }
    }
    *(u32x4*)&A1[e * 264 + c * 8] = val;
  }
  __syncthreads();

  const int lane = tid & 31;
  const int wv   = tid >> 5;
  const int nl   = lane & 15;
  const int half = lane >> 4;
  const int koff = half * 8;
  const int col  = wv * 16 + nl;

  v8f acc[4];
  {
    float bias = b1[col];
#pragma unroll
    for (int r = 0; r < 4; ++r)
#pragma unroll
      for (int i = 0; i < 8; ++i) acc[r][i] = bias;
  }

#pragma unroll
  for (int kk = 0; kk < 8; ++kk) {
    const unsigned short* bp = W1p + ((kk * DH + col) * 32 + half * 8);
    v16bf bf = frag2(bp, bp + 16);
#pragma unroll
    for (int r = 0; r < 4; ++r) {
      const unsigned short* ap = &A1[(r * 16 + nl) * 264 + kk * 32 + koff];
      v16bf af = frag2(ap, ap + 16);
      acc[r] = __builtin_amdgcn_wmma_f32_16x16x32_bf16(false, af, false, bf,
                                                       (short)0, acc[r], false, false);
    }
  }

#pragma unroll
  for (int r = 0; r < 4; ++r)
#pragma unroll
    for (int i = 0; i < 8; ++i) {
      int m = r * 16 + half * 8 + i;
      A2[m * 136 + col] = f2bf(fmaxf(acc[r][i], 0.f));
    }
  __syncthreads();

  v8f acc2[4];
  {
    float bias = b2[col];
#pragma unroll
    for (int r = 0; r < 4; ++r)
#pragma unroll
      for (int i = 0; i < 8; ++i) acc2[r][i] = bias;
  }

#pragma unroll
  for (int kk = 0; kk < 4; ++kk) {
    const unsigned short* bp = W2p + ((kk * DH + col) * 32 + half * 8);
    v16bf bf = frag2(bp, bp + 16);
#pragma unroll
    for (int r = 0; r < 4; ++r) {
      const unsigned short* ap = &A2[(r * 16 + nl) * 136 + kk * 32 + koff];
      v16bf af = frag2(ap, ap + 16);
      acc2[r] = __builtin_amdgcn_wmma_f32_16x16x32_bf16(false, af, false, bf,
                                                        (short)0, acc2[r], false, false);
    }
  }

#pragma unroll
  for (int r = 0; r < 4; ++r)
#pragma unroll
    for (int i = 0; i < 8; ++i) {
      int m = r * 16 + half * 8 + i;
      int node = n0 + m;
      if (node < N) {
        size_t idx = (size_t)node * DH + col;
        float nv = h_f32[idx] + acc2[r][i];
        h_f32[idx] = nv;
        h_bf[idx]  = f2bf(nv);
      }
    }
}

// ---------------------------------------------------------------------------
// Head: concat[h[u],h[v],rc] (257->pad288) -> relu 128 (WMMA) -> dot W2 (VALU).
// ---------------------------------------------------------------------------
__global__ __launch_bounds__(256) void head_kernel(const unsigned short* __restrict__ h_bf,
                                                   const float* __restrict__ coords,
                                                   const int* __restrict__ pairs,
                                                   const unsigned short* __restrict__ W1p,
                                                   const float* __restrict__ b1,
                                                   const float* __restrict__ W2,
                                                   const float* __restrict__ b2,
                                                   float* __restrict__ out,
                                                   int P) {
  __shared__ alignas(16) unsigned short A1[64 * 296];
  __shared__ alignas(16) float H2[64 * 136];
  __shared__ int   s_u[64];
  __shared__ int   s_v[64];
  __shared__ float s_rc[64];

  const int tid = threadIdx.x;
  const int p0  = blockIdx.x * 64;

  if (tid < 64) {
    int e = p0 + tid;
    int u = 0, v = 0; float rc = 0.f;
    if (e < P) {
      u = pairs[2 * e]; v = pairs[2 * e + 1];
      float dx = coords[2 * u]     - coords[2 * v];
      float dy = coords[2 * u + 1] - coords[2 * v + 1];
      rc = sqrtf(dx * dx + dy * dy + 1e-8f);
    }
    s_u[tid] = u; s_v[tid] = v; s_rc[tid] = rc;
  }
  __syncthreads();

  for (int idx = tid; idx < 64 * 36; idx += 256) {
    int e = idx / 36;
    int c = idx - e * 36;
    u32x4 val = {0u, 0u, 0u, 0u};
    if (p0 + e < P) {
      if (c < 16) {
        val = *(const u32x4*)(h_bf + (size_t)s_u[e] * DH + c * 8);
      } else if (c < 32) {
        val = *(const u32x4*)(h_bf + (size_t)s_v[e] * DH + (c - 16) * 8);
      } else if (c == 32) {
        val.x = (unsigned)f2bf(s_rc[e]);  // element 257 == rc, rest zero pad
      }
    }
    *(u32x4*)&A1[e * 296 + c * 8] = val;
  }
  __syncthreads();

  const int lane = tid & 31;
  const int wv   = tid >> 5;
  const int nl   = lane & 15;
  const int half = lane >> 4;
  const int koff = half * 8;
  const int col  = wv * 16 + nl;

  v8f acc[4];
  {
    float bias = b1[col];
#pragma unroll
    for (int r = 0; r < 4; ++r)
#pragma unroll
      for (int i = 0; i < 8; ++i) acc[r][i] = bias;
  }

#pragma unroll
  for (int kk = 0; kk < 9; ++kk) {
    const unsigned short* bp = W1p + ((kk * DH + col) * 32 + half * 8);
    v16bf bf = frag2(bp, bp + 16);
#pragma unroll
    for (int r = 0; r < 4; ++r) {
      const unsigned short* ap = &A1[(r * 16 + nl) * 296 + kk * 32 + koff];
      v16bf af = frag2(ap, ap + 16);
      acc[r] = __builtin_amdgcn_wmma_f32_16x16x32_bf16(false, af, false, bf,
                                                       (short)0, acc[r], false, false);
    }
  }

#pragma unroll
  for (int r = 0; r < 4; ++r)
#pragma unroll
    for (int i = 0; i < 8; ++i) {
      int m = r * 16 + half * 8 + i;
      H2[m * 136 + col] = fmaxf(acc[r][i], 0.f);
    }
  __syncthreads();

  if (tid < 64) {
    int e = p0 + tid;
    if (e < P) {
      float a = b2[0];
#pragma unroll 8
      for (int k = 0; k < DH; ++k) a = fmaf(H2[tid * 136 + k], W2[k], a);
      out[e] = a;
    }
  }
}

// ---------------------------------------------------------------------------
extern "C" void kernel_launch(void* const* d_in, const int* in_sizes, int n_in,
                              void* d_out, int out_size, void* d_ws, size_t ws_size,
                              hipStream_t stream) {
  (void)n_in; (void)out_size; (void)ws_size;

  const float* coords = (const float*)d_in[0];
  const int*   eidx   = (const int*)d_in[1];
  const int*   pairs  = (const int*)d_in[2];
  const float* nW1 = (const float*)d_in[3];
  const float* nb1 = (const float*)d_in[4];
  const float* nW2 = (const float*)d_in[5];
  const float* nb2 = (const float*)d_in[6];
  const float* mW1 = (const float*)d_in[7];
  const float* mb1 = (const float*)d_in[8];
  const float* mW2 = (const float*)d_in[9];
  const float* mb2 = (const float*)d_in[10];
  const float* uW1 = (const float*)d_in[11];
  const float* ub1 = (const float*)d_in[12];
  const float* uW2 = (const float*)d_in[13];
  const float* ub2 = (const float*)d_in[14];
  const float* hW1 = (const float*)d_in[15];
  const float* hb1 = (const float*)d_in[16];
  const float* hW2 = (const float*)d_in[17];
  const float* hb2 = (const float*)d_in[18];

  const int N = in_sizes[0] / 2;
  const int E = in_sizes[1] / 2;
  const int P = in_sizes[2] / 2;

  // Workspace layout (≈65 MB).
  float*          h_f32   = (float*)d_ws;
  float*          agg     = h_f32 + (size_t)N * DH;
  unsigned short* h_bf    = (unsigned short*)(agg + (size_t)N * DH);
  unsigned short* msgW1p  = h_bf + (size_t)N * DH;
  unsigned short* msgW2p  = msgW1p + (size_t)NL * 9 * DH * 32;
  unsigned short* updW1p  = msgW2p + (size_t)NL * 4 * DH * 32;
  unsigned short* updW2p  = updW1p + (size_t)NL * 8 * DH * 32;
  unsigned short* headW1p = updW2p + (size_t)NL * 4 * DH * 32;

  // Pack weights into WMMA B-fragment layout (bf16).
  for (int l = 0; l < NL; ++l) {
    pack_kernel<<<(9 * DH * 32 + 255) / 256, 256, 0, stream>>>(
        mW1 + (size_t)l * 272 * DH, msgW1p + (size_t)l * 9 * DH * 32, 272, 9 * DH * 32);
    pack_kernel<<<(4 * DH * 32 + 255) / 256, 256, 0, stream>>>(
        mW2 + (size_t)l * DH * DH, msgW2p + (size_t)l * 4 * DH * 32, 128, 4 * DH * 32);
    pack_kernel<<<(8 * DH * 32 + 255) / 256, 256, 0, stream>>>(
        uW1 + (size_t)l * 256 * DH, updW1p + (size_t)l * 8 * DH * 32, 256, 8 * DH * 32);
    pack_kernel<<<(4 * DH * 32 + 255) / 256, 256, 0, stream>>>(
        uW2 + (size_t)l * DH * DH, updW2p + (size_t)l * 4 * DH * 32, 128, 4 * DH * 32);
  }
  pack_kernel<<<(9 * DH * 32 + 255) / 256, 256, 0, stream>>>(hW1, headW1p, 257, 9 * DH * 32);

  node_kernel<<<N, 128, 0, stream>>>(coords, nW1, nb1, nW2, nb2, h_f32, h_bf, N);

  const int* srcI = eidx;
  const int* dstI = eidx + E;
  for (int l = 0; l < NL; ++l) {
    zero_kernel<<<(N * DH + 255) / 256, 256, 0, stream>>>(agg, N * DH);
    msg_kernel<<<(E + 63) / 64, 256, 0, stream>>>(
        h_bf, coords, srcI, dstI,
        msgW1p + (size_t)l * 9 * DH * 32, mb1 + l * DH,
        msgW2p + (size_t)l * 4 * DH * 32, mb2 + l * DH, agg, E);
    upd_kernel<<<(N + 63) / 64, 256, 0, stream>>>(
        h_f32, h_bf, agg,
        updW1p + (size_t)l * 8 * DH * 32, ub1 + l * DH,
        updW2p + (size_t)l * 4 * DH * 32, ub2 + l * DH, N);
  }

  head_kernel<<<(P + 63) / 64, 256, 0, stream>>>(
      h_bf, coords, pairs, headW1p, hb1, hW2, hb2, (float*)d_out, P);
}
